// DAWN_10419590660472
// MI455X (gfx1250) — matrix operand order repaired and checked by reading, not compiled, not run on hardware
//
#include <hip/hip_runtime.h>
#include <hip/hip_bf16.h>
#include <math.h>

// ---------------------------------------------------------------------------
// MI455X / gfx1250, wave32. All matmuls via v_wmma_f32_16x16x32_bf16.
// Weights converted f32->bf16 once per launch; all WMMA fragments are
// contiguous b128 loads (basis_A pre-transposed, V stored [B,H,DH,S]).
// GEMM: 32x64 tile per wave (8 WMMA / k-step); neuron kernel caches the
// full A strip (16 fragments) in registers across the 32-basis loop.
// ---------------------------------------------------------------------------

typedef __bf16 bf16_t;
typedef bf16_t v16bf __attribute__((ext_vector_type(16)));
typedef bf16_t v8bf  __attribute__((ext_vector_type(8)));
typedef float  v8f   __attribute__((ext_vector_type(8)));

#define DD    512
#define DFFN  2048
#define HH    8
#define DH    64
#define NNEU  256
#define NBAS  32
#define RRANK 128
#define KTOP  8
#define SS    1024
#define BBAT  4
#define LLAY  4
#define MTOK  (BBAT * SS)   // 4096 tokens
#define VOCN  32000

__device__ __forceinline__ v8f wmma_bf16(v16bf a, v16bf b, v8f c) {
  return __builtin_amdgcn_wmma_f32_16x16x32_bf16(false, a, false, b, (short)0, c,
                                                 false, false);
}

__device__ __forceinline__ v16bf cat8(v8bf lo, v8bf hi) {
  return __builtin_shufflevector(lo, hi, 0, 1, 2, 3, 4, 5, 6, 7, 8, 9, 10, 11, 12,
                                 13, 14, 15);
}

// A fragment 16x32 (MxK) from bf16 row-major; lane m=lane&15, half=lane>>4,
// elem e -> k = (e<8?0:16) + half*8 + (e&7).  Two contiguous 16B chunks.
__device__ __forceinline__ v16bf load_a_frag(const bf16_t* __restrict__ A, int lda,
                                             int lane) {
  const int m = lane & 15, half = lane >> 4;
  const bf16_t* r = A + (size_t)m * lda + half * 8;
  return cat8(*(const v8bf*)r, *(const v8bf*)(r + 16));
}

// B fragment from W stored [N,K] row-major bf16 (C = A * W^T); lane n=lane&15,
// half=lane>>4, elem e -> k = (e<8?0:16) + half*8 + (e&7).  Two contiguous 16B chunks.
__device__ __forceinline__ v16bf load_bt_frag(const bf16_t* __restrict__ W, int ldw,
                                              int lane) {
  const int n = lane & 15, half = lane >> 4;
  const bf16_t* r = W + (size_t)n * ldw + half * 8;
  return cat8(*(const v8bf*)r, *(const v8bf*)(r + 16));
}

// ---------------------------------------------------------------------------
// Generic GEMM: C[M,N] = epi(A[M,K](bf16) * W[N,K](bf16)^T + bias) (+ resid)
// OMODE: 0 = f32 out, 1 = bf16 out, 2 = bf16 out transposed to [B,H,DH,S]
// block = 128 threads = 4 waves; wave w: rows [by*128+w*32,+32), cols [bx*64,+64)
// 8 WMMAs per k-step per wave (2 A frags x 4 B frags).
// ---------------------------------------------------------------------------
template <bool BIAS, bool RESID, bool GELU, int OMODE>
__global__ __launch_bounds__(128) void gemm_kernel(
    const bf16_t* __restrict__ A, const bf16_t* __restrict__ W,
    const float* __restrict__ bias, const float* __restrict__ resid,
    void* __restrict__ Cout, int M, int N, int Kd) {
  const int lane = threadIdx.x & 31;
  const int wave = threadIdx.x >> 5;
  const int m0 = blockIdx.y * 128 + wave * 32;
  const int n0 = blockIdx.x * 64;
  const bf16_t* Ap = A + (size_t)m0 * Kd;
  const bf16_t* Wp = W + (size_t)n0 * Kd;
  v8f acc[2][4] = {};
  for (int k0 = 0; k0 < Kd; k0 += 32) {
    __builtin_prefetch(Wp + k0 + 128, 0, 1);  // global_prefetch next weight tile
    v16bf a0 = load_a_frag(Ap + k0, Kd, lane);
    v16bf a1 = load_a_frag(Ap + (size_t)16 * Kd + k0, Kd, lane);
#pragma unroll
    for (int t = 0; t < 4; ++t) {
      v16bf bt = load_bt_frag(Wp + (size_t)(t * 16) * Kd + k0, Kd, lane);
      acc[0][t] = wmma_bf16(a0, bt, acc[0][t]);
      acc[1][t] = wmma_bf16(a1, bt, acc[1][t]);
    }
  }
  const int nc = lane & 15, half = lane >> 4;
#pragma unroll
  for (int r = 0; r < 2; ++r) {
#pragma unroll
    for (int t = 0; t < 4; ++t) {
      const int n = n0 + t * 16 + nc;
      const float bv = BIAS ? bias[n] : 0.0f;
#pragma unroll
      for (int i = 0; i < 8; ++i) {
        const int m = m0 + r * 16 + i + half * 8;
        float v = acc[r][t][i] + bv;
        if (GELU) v = 0.5f * v * (1.0f + erff(v * 0.70710678118654752f));
        if (RESID) v += resid[(size_t)m * N + n];
        if (OMODE == 0) {
          ((float*)Cout)[(size_t)m * N + n] = v;
        } else if (OMODE == 1) {
          ((bf16_t*)Cout)[(size_t)m * N + n] = (bf16_t)v;
        } else {  // V-transpose: m = b*1024+s, n = h*64+dh -> [(b*8+h)*64+dh][s]
          const int bI = m >> 10, sI = m & (SS - 1);
          ((bf16_t*)Cout)[(size_t)(bI * (HH * DH) + n) * SS + sI] = (bf16_t)v;
        }
      }
    }
  }
}

// ---------------------------------------------------------------------------
// v_sem[m, r] = sum_nb tokrec[m,nb] * (normed[m,:] @ basis_A[nb,:,:])[r]
// basis_At is pre-transposed [nb][r][d] bf16 -> contiguous B fragments.
// One block per 16-row strip; 4 waves cover R=128 (32 cols each).
// The 16 A fragments (full K=512 strip) are cached in registers across nb.
// ---------------------------------------------------------------------------
__global__ __launch_bounds__(128) void neuron_vsem_kernel(
    const bf16_t* __restrict__ normed, const bf16_t* __restrict__ basisAt,
    const float* __restrict__ tokrec, bf16_t* __restrict__ vsem) {
  const int lane = threadIdx.x & 31;
  const int wave = threadIdx.x >> 5;
  const int m0 = blockIdx.x * 16;
  const int nc0 = wave * 32;
  const int half = lane >> 4;
  v16bf afr[16];
#pragma unroll
  for (int kk = 0; kk < 16; ++kk)
    afr[kk] = load_a_frag(normed + (size_t)m0 * DD + kk * 32, DD, lane);
  v8f out0 = {}, out1 = {};
  for (int nb = 0; nb < NBAS; ++nb) {
    const bf16_t* An = basisAt + (size_t)nb * RRANK * DD;  // [R, D] row-major
    v8f p0 = {}, p1 = {};
#pragma unroll
    for (int kk = 0; kk < 16; ++kk) {
      p0 = wmma_bf16(afr[kk], load_bt_frag(An + (size_t)nc0 * DD + kk * 32, DD, lane),
                     p0);
      p1 = wmma_bf16(afr[kk],
                     load_bt_frag(An + (size_t)(nc0 + 16) * DD + kk * 32, DD, lane),
                     p1);
    }
#pragma unroll
    for (int i = 0; i < 8; ++i) {
      const float trv = tokrec[(size_t)(m0 + i + half * 8) * NBAS + nb];
      out0[i] += trv * p0[i];
      out1[i] += trv * p1[i];
    }
  }
  const int nc = lane & 15;
#pragma unroll
  for (int i = 0; i < 8; ++i) {
    const int m = m0 + i + half * 8;
    vsem[(size_t)m * RRANK + nc0 + nc] = (bf16_t)out0[i];
    vsem[(size_t)m * RRANK + nc0 + 16 + nc] = (bf16_t)out1[i];
  }
}

// ---------------------------------------------------------------------------
// Flash attention: one wave per 16-query strip per (b,h).
// Q,K bf16 [B,S,D] (col = h*64+dh); V bf16 transposed [B,H,DH,S].
// Scores and P*V via WMMA; P staged C-frag -> bf16 LDS -> A-frag.
// ---------------------------------------------------------------------------
__global__ __launch_bounds__(128) void attention_kernel(
    const bf16_t* __restrict__ q, const bf16_t* __restrict__ k,
    const bf16_t* __restrict__ vt, bf16_t* __restrict__ ao) {
  __shared__ __align__(16) bf16_t plds[4][16 * 32];
  const int lane = threadIdx.x & 31;
  const int wave = threadIdx.x >> 5;
  const int h = blockIdx.y, b = blockIdx.z;
  const int q0 = (blockIdx.x * 4 + wave) * 16;
  const int half = lane >> 4, nc = lane & 15;
  const bf16_t* Qp = q + (size_t)(b * SS + q0) * DD + h * DH;
  const bf16_t* Kp = k + (size_t)b * SS * DD + h * DH;
  const bf16_t* Vp = vt + (size_t)(b * HH + h) * DH * SS;  // [DH][S]
  float rmax[8], rsum[8];
#pragma unroll
  for (int i = 0; i < 8; ++i) { rmax[i] = -INFINITY; rsum[i] = 0.0f; }
  v8f o0 = {}, o1 = {}, o2 = {}, o3 = {};
  const v16bf qa0 = load_a_frag(Qp, DD, lane);       // dh 0..31
  const v16bf qa1 = load_a_frag(Qp + 32, DD, lane);  // dh 32..63
  bf16_t* myp = plds[wave];
  for (int kt = 0; kt <= q0 + 15; kt += 32) {
    v8f s0 = {}, s1 = {};
    s0 = wmma_bf16(qa0, load_bt_frag(Kp + (size_t)kt * DD, DD, lane), s0);
    s0 = wmma_bf16(qa1, load_bt_frag(Kp + (size_t)kt * DD + 32, DD, lane), s0);
    s1 = wmma_bf16(qa0, load_bt_frag(Kp + (size_t)(kt + 16) * DD, DD, lane), s1);
    s1 = wmma_bf16(qa1, load_bt_frag(Kp + (size_t)(kt + 16) * DD + 32, DD, lane), s1);
#pragma unroll
    for (int i = 0; i < 8; ++i) {
      const int m = q0 + i + half * 8;
      float a0 = (kt + nc <= m) ? s0[i] * 0.125f : -INFINITY;
      float a1 = (kt + 16 + nc <= m) ? s1[i] * 0.125f : -INFINITY;
      float mx = fmaxf(a0, a1);
#pragma unroll
      for (int off = 8; off >= 1; off >>= 1) mx = fmaxf(mx, __shfl_xor(mx, off, 32));
      const float mnew = fmaxf(rmax[i], mx);
      const float scale = __expf(rmax[i] - mnew);
      rmax[i] = mnew;
      const float p0v = __expf(a0 - mnew);
      const float p1v = __expf(a1 - mnew);
      float ps = p0v + p1v;
#pragma unroll
      for (int off = 8; off >= 1; off >>= 1) ps += __shfl_xor(ps, off, 32);
      rsum[i] = rsum[i] * scale + ps;
      o0[i] *= scale; o1[i] *= scale; o2[i] *= scale; o3[i] *= scale;
      myp[(i + half * 8) * 32 + nc] = (bf16_t)p0v;
      myp[(i + half * 8) * 32 + 16 + nc] = (bf16_t)p1v;
    }
    asm volatile("s_wait_dscnt 0" ::: "memory");  // LDS RAW within wave
    v16bf pa;
    {
      const bf16_t* pr = myp + (lane & 15) * 32 + half * 8;
      pa = cat8(*(const v8bf*)pr, *(const v8bf*)(pr + 16));
    }
    // P * V : B[k=key][n=dh] = Vt[dh][kt+key], contiguous along keys
    o0 = wmma_bf16(pa, load_bt_frag(Vp + kt, SS, lane), o0);
    o1 = wmma_bf16(pa, load_bt_frag(Vp + (size_t)16 * SS + kt, SS, lane), o1);
    o2 = wmma_bf16(pa, load_bt_frag(Vp + (size_t)32 * SS + kt, SS, lane), o2);
    o3 = wmma_bf16(pa, load_bt_frag(Vp + (size_t)48 * SS + kt, SS, lane), o3);
  }
  bf16_t* aop = ao + (size_t)(b * SS + q0) * DD + h * DH;
#pragma unroll
  for (int i = 0; i < 8; ++i) {
    const float inv = 1.0f / rsum[i];
    const int mr = i + half * 8;
    aop[(size_t)mr * DD + nc] = (bf16_t)(o0[i] * inv);
    aop[(size_t)mr * DD + 16 + nc] = (bf16_t)(o1[i] * inv);
    aop[(size_t)mr * DD + 32 + nc] = (bf16_t)(o2[i] * inv);
    aop[(size_t)mr * DD + 48 + nc] = (bf16_t)(o3[i] * inv);
  }
}

// ---------------------------------------------------------------------------
// LayerNorm: one wave per token (D=512 -> 16 elems/lane); bf16 output.
// ---------------------------------------------------------------------------
__global__ __launch_bounds__(128) void layernorm_kernel(
    const float* __restrict__ x, const float* __restrict__ sc,
    const float* __restrict__ bt, bf16_t* __restrict__ out) {
  const int lane = threadIdx.x & 31;
  const int t = blockIdx.x * 4 + (threadIdx.x >> 5);
  const float* xp = x + (size_t)t * DD;
  float vals[16];
  float sum = 0.0f;
#pragma unroll
  for (int j = 0; j < 16; ++j) { vals[j] = xp[lane + 32 * j]; sum += vals[j]; }
#pragma unroll
  for (int off = 16; off >= 1; off >>= 1) sum += __shfl_xor(sum, off, 32);
  const float mean = sum * (1.0f / DD);
  float var = 0.0f;
#pragma unroll
  for (int j = 0; j < 16; ++j) { const float d = vals[j] - mean; var += d * d; }
#pragma unroll
  for (int off = 16; off >= 1; off >>= 1) var += __shfl_xor(var, off, 32);
  const float rstd = rsqrtf(var * (1.0f / DD) + 1e-5f);
  bf16_t* op = out + (size_t)t * DD;
#pragma unroll
  for (int j = 0; j < 16; ++j) {
    const int d = lane + 32 * j;
    op[d] = (bf16_t)((vals[j] - mean) * rstd * sc[d] + bt[d]);
  }
}

__global__ void embed_kernel(const int* __restrict__ ids,
                             const float* __restrict__ tok,
                             const float* __restrict__ pos,
                             float* __restrict__ x) {
  const int idx = blockIdx.x * blockDim.x + threadIdx.x;
  if (idx >= MTOK * DD) return;
  const int t = idx / DD, d = idx - t * DD;
  const int s = t & (SS - 1);
  x[idx] = tok[(size_t)ids[t] * DD + d] + pos[(size_t)s * DD + d];
}

// f32 -> bf16 bulk convert (n % 4 == 0)
__global__ void cvt_bf16_kernel(const float* __restrict__ src,
                                bf16_t* __restrict__ dst, int n) {
  const int i = (blockIdx.x * blockDim.x + threadIdx.x) * 4;
  if (i >= n) return;
  const float4 f = *(const float4*)(src + i);
  dst[i] = (bf16_t)f.x;
  dst[i + 1] = (bf16_t)f.y;
  dst[i + 2] = (bf16_t)f.z;
  dst[i + 3] = (bf16_t)f.w;
}

// basis_A[nb][d][r] (f32) -> basis_At[nb][r][d] (bf16)
__global__ void basis_transpose_kernel(const float* __restrict__ A,
                                       bf16_t* __restrict__ At) {
  const int idx = blockIdx.x * blockDim.x + threadIdx.x;
  if (idx >= NBAS * DD * RRANK) return;
  const int nb = idx >> 16;           // / (512*128)
  const int rem = idx & 65535;
  const int d = rem >> 7;             // / 128
  const int r = rem & 127;
  At[(size_t)nb * 65536 + r * DD + d] = (bf16_t)A[idx];
}

// emb_sem[n,:] = softmax(recipe_l[n,:]) @ basis_emb (bf16 out);
// sigc[n] = sigmoid(sum_h ctx_pat[n,h] / S)   (attn.mean(-1) == 1/S exactly)
__global__ __launch_bounds__(64) void embsem_kernel(
    const float* __restrict__ recipe_l, const float* __restrict__ ctx_l,
    const float* __restrict__ basis_emb, bf16_t* __restrict__ emb_sem,
    float* __restrict__ sigc) {
  const int n = blockIdx.x;
  float p[NBAS];
  float mx = -INFINITY;
#pragma unroll
  for (int j = 0; j < NBAS; ++j) { p[j] = recipe_l[n * NBAS + j]; mx = fmaxf(mx, p[j]); }
  float se = 0.0f;
#pragma unroll
  for (int j = 0; j < NBAS; ++j) { p[j] = __expf(p[j] - mx); se += p[j]; }
  const float inv = 1.0f / se;
  for (int d = threadIdx.x; d < DD; d += 64) {
    float acc = 0.0f;
#pragma unroll
    for (int j = 0; j < NBAS; ++j) acc += p[j] * basis_emb[(size_t)j * DD + d];
    emb_sem[(size_t)n * DD + d] = (bf16_t)(acc * inv);
  }
  if (threadIdx.x == 0) {
    float cs = 0.0f;
#pragma unroll
    for (int hh = 0; hh < HH; ++hh) cs += ctx_l[n * HH + hh];
    cs *= (1.0f / SS);
    sigc[n] = 1.0f / (1.0f + __expf(-cs));
  }
}

// Per token (one wave): fin = sem * sigc; top-8; w = softmax(topv);
// token_recipe[nb] = sum_k w_k * softmax(recipe[idx_k])[nb]   (lane <-> nb)
__global__ __launch_bounds__(128) void topk_kernel(
    const float* __restrict__ sem, const float* __restrict__ sigc,
    const float* __restrict__ recipe_l, float* __restrict__ tokrec) {
  const int lane = threadIdx.x & 31;
  const int t = blockIdx.x * 4 + (threadIdx.x >> 5);
  float fin[8];
#pragma unroll
  for (int j = 0; j < 8; ++j) {
    const int n = lane + 32 * j;
    fin[j] = sem[(size_t)t * NNEU + n] * sigc[n];
  }
  float topv[KTOP]; int topi[KTOP];
  for (int kk = 0; kk < KTOP; ++kk) {
    float bv = fin[0]; int bi = lane;
#pragma unroll
    for (int j = 1; j < 8; ++j)
      if (fin[j] > bv) { bv = fin[j]; bi = lane + 32 * j; }
#pragma unroll
    for (int off = 16; off >= 1; off >>= 1) {
      const float ov = __shfl_xor(bv, off, 32);
      const int oi = __shfl_xor(bi, off, 32);
      if (ov > bv || (ov == bv && oi < bi)) { bv = ov; bi = oi; }
    }
    topv[kk] = bv; topi[kk] = bi;
    if ((bi & 31) == lane) fin[bi >> 5] = -INFINITY;
  }
  float wmax = topv[0];
#pragma unroll
  for (int kk = 1; kk < KTOP; ++kk) wmax = fmaxf(wmax, topv[kk]);
  float w[KTOP]; float wsum = 0.0f;
#pragma unroll
  for (int kk = 0; kk < KTOP; ++kk) { w[kk] = __expf(topv[kk] - wmax); wsum += w[kk]; }
  const float winv = 1.0f / wsum;
  float acc = 0.0f;
#pragma unroll
  for (int kk = 0; kk < KTOP; ++kk) {
    const float rv = recipe_l[(size_t)topi[kk] * NBAS + lane];
    float rmx = rv;
#pragma unroll
    for (int off = 16; off >= 1; off >>= 1) rmx = fmaxf(rmx, __shfl_xor(rmx, off, 32));
    const float e = __expf(rv - rmx);
    float esum = e;
#pragma unroll
    for (int off = 16; off >= 1; off >>= 1) esum += __shfl_xor(esum, off, 32);
    acc += (w[kk] * winv) * (e / esum);
  }
  tokrec[(size_t)t * NBAS + lane] = acc;
}

// ---------------------------------------------------------------------------
extern "C" void kernel_launch(void* const* d_in, const int* in_sizes, int n_in,
                              void* d_out, int out_size, void* d_ws, size_t ws_size,
                              hipStream_t stream) {
  const int* ids = (const int*)d_in[0];
  const float* tok_emb = (const float*)d_in[1];
  const float* pos_emb = (const float*)d_in[2];
  const float* basis_A = (const float*)d_in[3];
  const float* basis_emb = (const float*)d_in[4];
  const float* q_w = (const float*)d_in[5];
  const float* q_b = (const float*)d_in[6];
  const float* k_w = (const float*)d_in[7];
  const float* k_b = (const float*)d_in[8];
  const float* ao_w = (const float*)d_in[9];
  const float* ao_b = (const float*)d_in[10];
  const float* recipe = (const float*)d_in[11];
  const float* ctx_pat = (const float*)d_in[12];
  const float* vout_w = (const float*)d_in[13];
  const float* vout_b = (const float*)d_in[14];
  const float* up_w = (const float*)d_in[15];
  const float* up_b = (const float*)d_in[16];
  const float* down_w = (const float*)d_in[17];
  const float* down_b = (const float*)d_in[18];
  const float* ln1_s = (const float*)d_in[19];
  const float* ln1_b = (const float*)d_in[20];
  const float* ln2_s = (const float*)d_in[21];
  const float* ln2_b = (const float*)d_in[22];
  const float* lnf_s = (const float*)d_in[23];
  const float* lnf_b = (const float*)d_in[24];
  float* out = (float*)d_out;

  // ---- workspace carve-up: f32 region then bf16 region (~96 MB total) ----
  float* ws = (float*)d_ws;
  float* x = ws;                                   // [4096,512] f32
  float* sem = x + (size_t)MTOK * DD;              // [4096,256] f32
  float* trbuf = sem + (size_t)MTOK * NNEU;        // [4096,32]  f32
  float* sigc = trbuf + (size_t)MTOK * NBAS;       // [256]      f32
  bf16_t* bbase = (bf16_t*)(sigc + 256);
  bf16_t* normed = bbase;                          // [4096,512]
  bf16_t* qbf = normed + (size_t)MTOK * DD;        // [4096,512]
  bf16_t* kbf = qbf + (size_t)MTOK * DD;           // [4096,512]
  bf16_t* vtbf = kbf + (size_t)MTOK * DD;          // [B,H,DH,S] = [4096,512]
  bf16_t* aobf = vtbf + (size_t)MTOK * DD;         // [4096,512]
  bf16_t* hbf = qbf;                               // [4096,2048] aliases q/k/vt/ao
  bf16_t* vsem = aobf + (size_t)MTOK * DD;         // [4096,128]
  bf16_t* embsem = vsem + (size_t)MTOK * RRANK;    // [256,512]
  bf16_t* wq = embsem + (size_t)NNEU * DD;         // [4,512,512]
  bf16_t* wk = wq + (size_t)LLAY * DD * DD;
  bf16_t* wao = wk + (size_t)LLAY * DD * DD;
  bf16_t* wvout = wao + (size_t)LLAY * DD * DD;    // [4,512,128]
  bf16_t* wup = wvout + (size_t)LLAY * DD * RRANK; // [4,2048,512]
  bf16_t* wdown = wup + (size_t)LLAY * DFFN * DD;  // [4,512,2048]
  bf16_t* wtok = wdown + (size_t)LLAY * DD * DFFN; // [32000,512]
  bf16_t* wbasT = wtok + (size_t)VOCN * DD;        // [32,128,512]

  const dim3 blk(128);
  auto cvt = [&](const float* s, bf16_t* d, int n) {
    cvt_bf16_kernel<<<dim3((n / 4 + 255) / 256), dim3(256), 0, stream>>>(s, d, n);
  };
  // one-time (per launch) weight conversion; everything stays L2-resident after
  cvt(q_w, wq, LLAY * DD * DD);
  cvt(k_w, wk, LLAY * DD * DD);
  cvt(ao_w, wao, LLAY * DD * DD);
  cvt(vout_w, wvout, LLAY * DD * RRANK);
  cvt(up_w, wup, LLAY * DFFN * DD);
  cvt(down_w, wdown, LLAY * DD * DFFN);
  cvt(tok_emb, wtok, VOCN * DD);
  basis_transpose_kernel<<<dim3((NBAS * DD * RRANK + 255) / 256), dim3(256), 0,
                           stream>>>(basis_A, wbasT);

  embed_kernel<<<dim3((MTOK * DD + 255) / 256), dim3(256), 0, stream>>>(ids, tok_emb,
                                                                        pos_emb, x);
  for (int l = 0; l < LLAY; ++l) {
    layernorm_kernel<<<dim3(MTOK / 4), blk, 0, stream>>>(x, ln1_s + l * DD,
                                                         ln1_b + l * DD, normed);
    gemm_kernel<true, false, false, 1><<<dim3(DD / 64, MTOK / 128), blk, 0, stream>>>(
        normed, wq + (size_t)l * DD * DD, q_b + l * DD, nullptr, qbf, MTOK, DD, DD);
    gemm_kernel<true, false, false, 1><<<dim3(DD / 64, MTOK / 128), blk, 0, stream>>>(
        normed, wk + (size_t)l * DD * DD, k_b + l * DD, nullptr, kbf, MTOK, DD, DD);
    embsem_kernel<<<dim3(NNEU), dim3(64), 0, stream>>>(
        recipe + (size_t)l * NNEU * NBAS, ctx_pat + (size_t)l * NNEU * HH, basis_emb,
        embsem, sigc);
    gemm_kernel<false, false, false, 0><<<dim3(NNEU / 64, MTOK / 128), blk, 0,
                                          stream>>>(normed, embsem, nullptr, nullptr,
                                                    sem, MTOK, NNEU, DD);
    topk_kernel<<<dim3(MTOK / 4), blk, 0, stream>>>(
        sem, sigc, recipe + (size_t)l * NNEU * NBAS, trbuf);
    neuron_vsem_kernel<<<dim3(MTOK / 16), blk, 0, stream>>>(normed, wbasT, trbuf,
                                                            vsem);
    gemm_kernel<true, false, false, 2><<<dim3(DD / 64, MTOK / 128), blk, 0, stream>>>(
        vsem, wvout + (size_t)l * DD * RRANK, vout_b + l * DD, nullptr, vtbf, MTOK,
        DD, RRANK);
    attention_kernel<<<dim3(SS / 64, HH, BBAT), blk, 0, stream>>>(qbf, kbf, vtbf,
                                                                  aobf);
    gemm_kernel<true, true, false, 0><<<dim3(DD / 64, MTOK / 128), blk, 0, stream>>>(
        aobf, wao + (size_t)l * DD * DD, ao_b + l * DD, x, x, MTOK, DD, DD);
    layernorm_kernel<<<dim3(MTOK / 4), blk, 0, stream>>>(x, ln2_s + l * DD,
                                                         ln2_b + l * DD, normed);
    gemm_kernel<true, false, true, 1><<<dim3(DFFN / 64, MTOK / 128), blk, 0,
                                        stream>>>(normed, wup + (size_t)l * DFFN * DD,
                                                  up_b + l * DFFN, nullptr, hbf, MTOK,
                                                  DFFN, DD);
    gemm_kernel<true, true, false, 0><<<dim3(DD / 64, MTOK / 128), blk, 0, stream>>>(
        hbf, wdown + (size_t)l * DD * DFFN, down_b + l * DD, x, x, MTOK, DD, DFFN);
  }
  layernorm_kernel<<<dim3(MTOK / 4), blk, 0, stream>>>(x, lnf_s, lnf_b, normed);
  gemm_kernel<false, false, false, 0><<<dim3(VOCN / 64, MTOK / 128), blk, 0,
                                        stream>>>(normed, wtok, nullptr, nullptr, out,
                                                  MTOK, VOCN, DD);
}
